// RopeCausalSelfAttention_69947837382830
// MI455X (gfx1250) — matrix operand order repaired and checked by reading, not compiled, not run on hardware
//
#include <hip/hip_runtime.h>
#include <hip/hip_bf16.h>
#include <math.h>

// ---------------------------------------------------------------------------
// RoPE + causal self-attention forward, CDNA5 (gfx1250, wave32).
// Matmuls: v_wmma_f32_16x16x32_f16 (64x64 wave tiles in the GEMMs).
// K/V attention tiles staged to LDS with the Tensor Data Mover
// (tensor_load_to_lds / TENSORcnt), double-buffered.
// Weights pre-swizzled to B-fragment-major layout (one 32B load per lane).
// ---------------------------------------------------------------------------

typedef __attribute__((ext_vector_type(16))) _Float16 v16h;
typedef __attribute__((ext_vector_type(8)))  _Float16 v8h;
typedef __attribute__((ext_vector_type(8)))  float    v8f;
typedef __attribute__((ext_vector_type(4)))  unsigned su4;
typedef __attribute__((ext_vector_type(8)))  unsigned su8;

#define WMMA_F16(a, b, c) \
  __builtin_amdgcn_wmma_f32_16x16x32_f16(false, (a), false, (b), (short)0, (c), false, false)

constexpr int BSZ   = 2;
constexpr int T     = 2048;
constexpr int C     = 1024;
constexpr int H     = 16;
constexpr int DK    = 64;            // C / H
constexpr int MROWS = BSZ * T;       // 4096
constexpr int NQKV  = 3 * C;         // 3072

// ---- workspace layout (bytes) ---------------------------------------------
constexpr size_t OFF_QKV  = 0;                                    // fp32 [4096][3072]
constexpr size_t OFF_X16  = OFF_QKV  + (size_t)MROWS * NQKV * 4;  // f16  [4096][1024]
constexpr size_t OFF_WQ16 = OFF_X16  + (size_t)MROWS * C * 2;     // f16  packed frags
constexpr size_t OFF_WP16 = OFF_WQ16 + (size_t)C * NQKV * 2;      // f16  packed frags
constexpr size_t OFF_Q    = OFF_WP16 + (size_t)C * C * 2;         // f16  [B,H,T,64]
constexpr size_t OFF_K    = OFF_Q    + (size_t)BSZ * H * T * DK * 2;
constexpr size_t OFF_V    = OFF_K    + (size_t)BSZ * H * T * DK * 2; // f16 [B,H,64,T]
constexpr size_t OFF_O    = OFF_V    + (size_t)BSZ * H * DK * T * 2; // f16 [B,T,C]

// ---------------------------------------------------------------------------
// TDM helpers: D# group0/group1 packing per CDNA5 ISA 8.3/8.4, 2-group form.
// ---------------------------------------------------------------------------
typedef __attribute__((address_space(3))) const void* lds_cvptr;
__device__ __forceinline__ unsigned lds_offset(const void* p) {
  return (unsigned)(unsigned long long)(lds_cvptr)p;   // LDS byte address
}

__device__ __forceinline__ su4 tdm_g0(unsigned lds_addr, unsigned long long gaddr) {
  su4 g;
  g[0] = 1u;                                            // count=1, user mode
  g[1] = lds_addr;                                      // lds_addr[31:0]
  g[2] = (unsigned)(gaddr & 0xFFFFFFFFu);               // global_addr[31:0]
  g[3] = (unsigned)((gaddr >> 32) & 0x1FFFFFFu) | (2u << 30);  // addr[56:32] | type=2
  return g;
}

// 2-D tile of 16-bit elements: X = tile_d0 contiguous, Y = tile_d1 rows,
// row stride = stride0 elements.  LDS destination is written contiguously.
__device__ __forceinline__ su8 tdm_g1_2d(unsigned tile_d0, unsigned tile_d1,
                                         unsigned long long stride0) {
  const unsigned td0 = 0x40000000u, td1 = 0x40000000u;  // huge dims: never OOB
  su8 g;
  g[0] = (1u << 16);                                    // data_size = 2 bytes
  g[1] = (td0 & 0xFFFFu) << 16;                         // tensor_dim0 lo16
  g[2] = (td0 >> 16) | ((td1 & 0xFFFFu) << 16);         // dim0 hi16 | dim1 lo16
  g[3] = (td1 >> 16) | (tile_d0 << 16);                 // dim1 hi16 | tile_dim0
  g[4] = tile_d1 & 0xFFFFu;                             // tile_dim1 (tile_dim2=0)
  g[5] = (unsigned)(stride0 & 0xFFFFFFFFu);             // dim0_stride lo32
  g[6] = (unsigned)((stride0 >> 32) & 0xFFFFu);         // dim0_stride hi16
  g[7] = 0;
  return g;
}

__device__ __forceinline__ void tdm_load(su4 g0, su8 g1) {
  asm volatile("tensor_load_to_lds %0, %1" :: "s"(g0), "s"(g1) : "memory");
}

// ---------------------------------------------------------------------------
// Fragment loaders (wave32, contiguous k order: k = 16*(lane/16) + s, same as packed B)
// ---------------------------------------------------------------------------
__device__ __forceinline__ v16h load_a_f16src(const _Float16* __restrict__ A, int lda,
                                              int row0, int k0, int lane) {
  const _Float16* p = A + (size_t)(row0 + (lane & 15)) * lda + k0 + ((lane >> 4) << 4);
  v8h lo = *(const v8h*)(p);
  v8h hi = *(const v8h*)(p + 8);
  v16h a;
#pragma unroll
  for (int i = 0; i < 8; ++i) { a[i] = lo[i]; a[8 + i] = hi[i]; }
  return a;
}

// packed fragment-major weights: frag = kt*ntilesN + nt, lane-contiguous 32B
__device__ __forceinline__ v16h load_b_packed(const _Float16* __restrict__ Wp,
                                              int ntilesN, int kt, int nt, int lane) {
  return *(const v16h*)(Wp + (((size_t)(kt * ntilesN + nt) * 32) + lane) * 16);
}

// ---------------------------------------------------------------------------
// Kernel 0a: fp32 -> f16 elementwise
// ---------------------------------------------------------------------------
__global__ __launch_bounds__(256) void k_f32_to_f16(const float* __restrict__ src,
                                                    _Float16* __restrict__ dst, int n) {
  int i = blockIdx.x * 256 + threadIdx.x;
  if (i < n) dst[i] = (_Float16)src[i];
}

// ---------------------------------------------------------------------------
// Kernel 0b: pack row-major fp32 W[K][N] -> f16 B-fragment-major layout
// ---------------------------------------------------------------------------
__global__ __launch_bounds__(256) void k_pack_b(const float* __restrict__ W,
                                                _Float16* __restrict__ dst,
                                                int N, int nfrag) {
  int tid = blockIdx.x * 256 + threadIdx.x;
  int lane = tid & 31;
  int frag = tid >> 5;
  if (frag >= nfrag) return;
  int ntilesN = N >> 4;
  int nt = frag % ntilesN;
  int kt = frag / ntilesN;
  const float* src = W + (size_t)(kt * 32 + ((lane >> 4) << 4)) * N + nt * 16 + (lane & 15);
  _Float16* d = dst + ((size_t)frag * 32 + lane) * 16;
#pragma unroll
  for (int i = 0; i < 16; ++i) d[i] = (_Float16)src[(size_t)i * N];
}

// ---------------------------------------------------------------------------
// Shared GEMM body: wave tile 64x64 (4x4 WMMA accumulators), block = 8 waves
// arranged 2(M) x 4(N) -> block tile 128 x 256.
// ---------------------------------------------------------------------------
__device__ __forceinline__ void gemm_wave64x64(const _Float16* __restrict__ A, int K,
                                               const _Float16* __restrict__ Wp, int ntilesN,
                                               const float* __restrict__ bias,
                                               float* __restrict__ out, int ldo,
                                               int row0, int col0, int lane) {
  const int nt0 = col0 >> 4;
  v8f acc[4][4] = {};
  for (int k = 0; k < K; k += 32) {
    if (k + 32 < K)
      __builtin_prefetch(A + (size_t)(row0 + (lane & 15)) * K + k + 32, 0, 1);
    v16h a[4];
#pragma unroll
    for (int i = 0; i < 4; ++i) a[i] = load_a_f16src(A, K, row0 + 16 * i, k, lane);
    const int kt = k >> 5;
#pragma unroll
    for (int j = 0; j < 4; ++j) {
      v16h b = load_b_packed(Wp, ntilesN, kt, nt0 + j, lane);
#pragma unroll
      for (int i = 0; i < 4; ++i) acc[i][j] = WMMA_F16(a[i], b, acc[i][j]);
    }
  }
  const int rsel = (lane >> 4) << 3;
  const int csel = lane & 15;
#pragma unroll
  for (int i = 0; i < 4; ++i)
#pragma unroll
    for (int j = 0; j < 4; ++j)
#pragma unroll
      for (int r = 0; r < 8; ++r) {
        int row = row0 + 16 * i + r + rsel;
        int col = col0 + 16 * j + csel;
        out[(size_t)row * ldo + col] = acc[i][j][r] + bias[col];
      }
}

// ---------------------------------------------------------------------------
// Kernel 1: QKV GEMM  qkv[4096][3072] = x16 @ Wq + b
// ---------------------------------------------------------------------------
__global__ __launch_bounds__(256) void k_qkv_gemm(const _Float16* __restrict__ x16,
                                                  const _Float16* __restrict__ Wp,
                                                  const float* __restrict__ bias,
                                                  float* __restrict__ qkv) {
  const int lane = threadIdx.x & 31;
  const int w    = threadIdx.x >> 5;
  const int row0 = blockIdx.y * 128 + (w & 1) * 64;
  const int col0 = blockIdx.x * 256 + (w >> 1) * 64;
  gemm_wave64x64(x16, C, Wp, NQKV >> 4, bias, qkv, NQKV, row0, col0, lane);
}

// ---------------------------------------------------------------------------
// Kernel 2: RoPE (interleaved) + repack.
//   Qh/Kh: f16 [B,H,T,64]   Vt: f16 [B,H,64,T]
// ---------------------------------------------------------------------------
__global__ __launch_bounds__(256) void k_rope_pack(const float* __restrict__ qkv,
                                                   _Float16* __restrict__ Qh,
                                                   _Float16* __restrict__ Kh,
                                                   _Float16* __restrict__ Vt) {
  int idx = blockIdx.x * 256 + threadIdx.x;   // [b:1][t:11][h:4][i:5]
  int i = idx & 31;
  int h = (idx >> 5) & (H - 1);
  int t = (idx >> 9) & (T - 1);
  int b = idx >> 20;

  size_t rowbase = (size_t)(b * T + t) * NQKV + h * DK;
  float qe = qkv[rowbase + 2 * i],     qo = qkv[rowbase + 2 * i + 1];
  float ke = qkv[rowbase + C + 2 * i], ko = qkv[rowbase + C + 2 * i + 1];
  float ve = qkv[rowbase + 2 * C + 2 * i], vo = qkv[rowbase + 2 * C + 2 * i + 1];

  float freq = __expf(-(float)(2 * i) * (9.2103403719761827f / 64.0f)); // ln(1e4)/64
  float ang  = (float)t * freq;
  float sn, cs;
  __sincosf(ang, &sn, &cs);

  size_t bh     = (size_t)(b * H + h);
  size_t qkbase = (bh * T + t) * DK;
  Qh[qkbase + 2 * i]     = (_Float16)(qe * cs - qo * sn);
  Qh[qkbase + 2 * i + 1] = (_Float16)(qe * sn + qo * cs);
  Kh[qkbase + 2 * i]     = (_Float16)(ke * cs - ko * sn);
  Kh[qkbase + 2 * i + 1] = (_Float16)(ke * sn + ko * cs);

  size_t vbase = bh * DK * T;
  Vt[vbase + (size_t)(2 * i) * T + t]     = (_Float16)ve;
  Vt[vbase + (size_t)(2 * i + 1) * T + t] = (_Float16)vo;
}

// ---------------------------------------------------------------------------
// Kernel 3: cooperative causal flash attention.
// Block = 8 waves = 128 query rows of one (b,h). K/V 32-key tiles are
// TDM-loaded into LDS (double-buffered) by wave 0; all waves consume them.
// ---------------------------------------------------------------------------
__global__ __launch_bounds__(256) void k_attn(const _Float16* __restrict__ Qh,
                                              const _Float16* __restrict__ Kh,
                                              const _Float16* __restrict__ Vt,
                                              _Float16* __restrict__ O) {
  __shared__ __align__(32) _Float16 kbuf[2][32 * 64];  // [key][d]
  __shared__ __align__(32) _Float16 vbuf[2][64 * 32];  // [d][key]
  __shared__ _Float16 plds[8][16][34];                 // per-wave P staging

  const int lane = threadIdx.x & 31;
  const int w    = threadIdx.x >> 5;
  const int bh   = blockIdx.y;
  const int qb0  = blockIdx.x * 128;
  const int q0   = qb0 + w * 16;

  const _Float16* Qb = Qh + (size_t)bh * T * DK;
  const _Float16* Kb = Kh + (size_t)bh * T * DK;
  const _Float16* Vb = Vt + (size_t)bh * DK * T;

  v16h qf0 = load_a_f16src(Qb, DK, q0, 0, lane);
  v16h qf1 = load_a_f16src(Qb, DK, q0, 32, lane);

  v8f   oacc[4] = {};
  float mprev[8], lsum[8];
#pragma unroll
  for (int r = 0; r < 8; ++r) { mprev[r] = -1e30f; lsum[r] = 0.0f; }

  const int rsel = (lane >> 4) << 3;
  const int csel = lane & 15;
  const int nkb  = (qb0 + 128) / 32;   // key blocks covering rows qb0..qb0+127

  // prologue: stage tile 0
  if (w == 0) {
    tdm_load(tdm_g0(lds_offset(&kbuf[0][0]), (unsigned long long)Kb),
             tdm_g1_2d(32 * 64, 1, 32 * 64));          // 1-D contiguous 4KB
    tdm_load(tdm_g0(lds_offset(&vbuf[0][0]), (unsigned long long)Vb),
             tdm_g1_2d(32, 64, T));                    // 64 rows x 32, stride T
    __builtin_amdgcn_s_wait_tensorcnt(0);
  }
  __syncthreads();

  for (int kb = 0; kb < nkb; ++kb) {
    const int kbase = kb * 32;
    const _Float16* kt = &kbuf[kb & 1][0];
    const _Float16* vt = &vbuf[kb & 1][0];

    // issue next tile while computing this one
    if (w == 0 && kb + 1 < nkb) {
      tdm_load(tdm_g0(lds_offset(&kbuf[(kb + 1) & 1][0]),
                      (unsigned long long)(Kb + (size_t)(kbase + 32) * DK)),
               tdm_g1_2d(32 * 64, 1, 32 * 64));
      tdm_load(tdm_g0(lds_offset(&vbuf[(kb + 1) & 1][0]),
                      (unsigned long long)(Vb + kbase + 32)),
               tdm_g1_2d(32, 64, T));
    }

    if (kbase <= q0 + 15) {   // causal: this wave has live keys in the tile
      // ---- S = Q K^T : B-frags from LDS K tile, contiguous 32B per lane ----
      v8f s0 = {}, s1 = {};
      {
        const _Float16* pk0 = kt + (csel)*DK + ((lane >> 4) << 4);
        const _Float16* pk1 = kt + (16 + csel) * DK + ((lane >> 4) << 4);
        v16h b00 = *(const v16h*)(pk0);
        v16h b01 = *(const v16h*)(pk0 + 32);
        v16h b10 = *(const v16h*)(pk1);
        v16h b11 = *(const v16h*)(pk1 + 32);
        s0 = WMMA_F16(qf0, b00, s0);
        s0 = WMMA_F16(qf1, b01, s0);
        s1 = WMMA_F16(qf0, b10, s1);
        s1 = WMMA_F16(qf1, b11, s1);
      }

      // ---- scale + causal mask + row max ----
      float mcur[8];
#pragma unroll
      for (int r = 0; r < 8; ++r) {
        int row = q0 + rsel + r;
        float v0 = s0[r] * 0.125f;               // 1/sqrt(64)
        float v1 = s1[r] * 0.125f;
        if (kbase + csel > row)      v0 = -1e30f;
        if (kbase + 16 + csel > row) v1 = -1e30f;
        s0[r] = v0; s1[r] = v1;
        mcur[r] = fmaxf(v0, v1);
      }
#pragma unroll
      for (int d = 1; d < 16; d <<= 1)
#pragma unroll
        for (int r = 0; r < 8; ++r)
          mcur[r] = fmaxf(mcur[r], __shfl_xor(mcur[r], d));

      // ---- online softmax ----
      float p0[8], p1[8];
#pragma unroll
      for (int r = 0; r < 8; ++r) {
        float mnew = fmaxf(mprev[r], mcur[r]);
        float corr = __expf(mprev[r] - mnew);
        p0[r] = __expf(s0[r] - mnew);
        p1[r] = __expf(s1[r] - mnew);
        float rs = p0[r] + p1[r];
#pragma unroll
        for (int d = 1; d < 16; d <<= 1) rs += __shfl_xor(rs, d);
        lsum[r]  = lsum[r] * corr + rs;
        mprev[r] = mnew;
#pragma unroll
        for (int dd = 0; dd < 4; ++dd) oacc[dd][r] *= corr;
      }

      // ---- C-layout P -> LDS -> A-frag layout ----
#pragma unroll
      for (int r = 0; r < 8; ++r) {
        plds[w][rsel + r][csel]      = (_Float16)p0[r];
        plds[w][rsel + r][16 + csel] = (_Float16)p1[r];
      }
      v16h pa;
      {
        int arow = lane & 15;
        int ak   = (lane >> 4) << 4;
#pragma unroll
        for (int i = 0; i < 8; ++i) {
          pa[i]     = plds[w][arow][ak + i];
          pa[8 + i] = plds[w][arow][ak + 8 + i];
        }
      }

      // ---- O += P V : B-frags from LDS V tile [64][32] ----
#pragma unroll
      for (int dd = 0; dd < 4; ++dd) {
        const _Float16* pv = vt + (dd * 16 + csel) * 32 + ((lane >> 4) << 4);
        v16h vf = *(const v16h*)pv;
        oacc[dd] = WMMA_F16(pa, vf, oacc[dd]);
      }
    }

    if (w == 0) __builtin_amdgcn_s_wait_tensorcnt(0);  // next tile landed
    __syncthreads();
  }

  // ---- normalize + write f16 O in [B,T,C] layout ----
  const int b = bh >> 4;
  const int h = bh & (H - 1);
#pragma unroll
  for (int dd = 0; dd < 4; ++dd)
#pragma unroll
    for (int r = 0; r < 8; ++r) {
      int t = q0 + rsel + r;
      size_t off = (size_t)(b * T + t) * C + h * DK + dd * 16 + csel;
      O[off] = (_Float16)(oacc[dd][r] / lsum[r]);
    }
}

// ---------------------------------------------------------------------------
// Kernel 4: output projection  out[4096][1024] = O @ Wp + b
// ---------------------------------------------------------------------------
__global__ __launch_bounds__(256) void k_proj_gemm(const _Float16* __restrict__ A,
                                                   const _Float16* __restrict__ Wp,
                                                   const float* __restrict__ bias,
                                                   float* __restrict__ out) {
  const int lane = threadIdx.x & 31;
  const int w    = threadIdx.x >> 5;
  const int row0 = blockIdx.y * 128 + (w & 1) * 64;
  const int col0 = blockIdx.x * 256 + (w >> 1) * 64;
  gemm_wave64x64(A, C, Wp, C >> 4, bias, out, C, row0, col0, lane);
}

// ---------------------------------------------------------------------------
extern "C" void kernel_launch(void* const* d_in, const int* in_sizes, int n_in,
                              void* d_out, int out_size, void* d_ws, size_t ws_size,
                              hipStream_t stream) {
  const float* x      = (const float*)d_in[0];
  const float* W_qkv  = (const float*)d_in[1];
  const float* b_qkv  = (const float*)d_in[2];
  const float* W_proj = (const float*)d_in[3];
  const float* b_proj = (const float*)d_in[4];

  char* ws = (char*)d_ws;
  float*     qkv  = (float*)(ws + OFF_QKV);
  _Float16*  X16  = (_Float16*)(ws + OFF_X16);
  _Float16*  Wq16 = (_Float16*)(ws + OFF_WQ16);
  _Float16*  Wp16 = (_Float16*)(ws + OFF_WP16);
  _Float16*  Qh   = (_Float16*)(ws + OFF_Q);
  _Float16*  Kh   = (_Float16*)(ws + OFF_K);
  _Float16*  Vt   = (_Float16*)(ws + OFF_V);
  _Float16*  Oh   = (_Float16*)(ws + OFF_O);

  // 0) precision conversion + weight fragment packing
  k_f32_to_f16<<<(MROWS * C + 255) / 256, 256, 0, stream>>>(x, X16, MROWS * C);
  {
    int nfragQ = (C / 32) * (NQKV / 16);       // 6144 frags
    k_pack_b<<<(nfragQ * 32 + 255) / 256, 256, 0, stream>>>(W_qkv, Wq16, NQKV, nfragQ);
    int nfragP = (C / 32) * (C / 16);          // 2048 frags
    k_pack_b<<<(nfragP * 32 + 255) / 256, 256, 0, stream>>>(W_proj, Wp16, C, nfragP);
  }

  // 1) QKV GEMM  (block tile 128 x 256)
  dim3 g1(NQKV / 256, MROWS / 128);
  k_qkv_gemm<<<g1, 256, 0, stream>>>(X16, Wq16, b_qkv, qkv);

  // 2) RoPE + repack
  k_rope_pack<<<(BSZ * T * H * 32) / 256, 256, 0, stream>>>(qkv, Qh, Kh, Vt);

  // 3) cooperative causal flash attention (TDM-staged K/V)
  dim3 g3(T / 128, BSZ * H);
  k_attn<<<g3, 256, 0, stream>>>(Qh, Kh, Vt, Oh);

  // 4) output projection  (block tile 128 x 256)
  dim3 g4(C / 256, MROWS / 128);
  k_proj_gemm<<<g4, 256, 0, stream>>>(Oh, Wp16, b_proj, (float*)d_out);
}
